// MultiHeadAttentionWithRelativePos_46145128628509
// MI455X (gfx1250) — matrix operand order repaired and checked by reading, not compiled, not run on hardware
//
#include <hip/hip_runtime.h>

typedef __bf16 bf16_t;
typedef __attribute__((ext_vector_type(16))) __bf16 v16bf;
typedef __attribute__((ext_vector_type(8)))  float  v8f;
typedef unsigned int v4u __attribute__((ext_vector_type(4)));
typedef int v4i __attribute__((ext_vector_type(4)));
typedef int v8i __attribute__((ext_vector_type(8)));

#define WMMA_BF16(a, b, c) \
  __builtin_amdgcn_wmma_f32_16x16x32_bf16(false, (a), false, (b), (short)0, (c), false, false)

// ---------------------------------------------------------------------------
// Problem constants
// ---------------------------------------------------------------------------
static constexpr int Bn  = 2;
static constexpr int Sn  = 2048;
static constexpr int Dn  = 1024;
static constexpr int Hn  = 16;
static constexpr int HDn = 64;
static constexpr int Mn  = Bn * Sn;   // 4096 rows for projection GEMMs

// ---------------------------------------------------------------------------
// Fragment loaders (wave32; A/B use contiguous k order, same in both; C/D as in ISA 7.12.2)
// A (16x32 bf16, MxK): lane l -> row m=l&15; elem e -> K=(l>=16?16:0)+e
// B (32x16 bf16, KxN) stored N-major (ld K-elems per row n):
//                      lane l -> col n=l&15; elem e -> K=(l>=16?16:0)+e
// C/D (16x16 f32): lane l -> col n=l&15; vgpr v -> row m=v+(l>=16)*8
// ---------------------------------------------------------------------------
__device__ __forceinline__ v16bf load_a_frag(const bf16_t* base, int ld, int lane) {
  const int m  = lane & 15;
  const int hh = lane >> 4;
  const bf16_t* p = base + (size_t)m * ld + hh * 16;
  v16bf a;
#pragma unroll
  for (int e = 0; e < 16; ++e) {
    a[e] = p[e];
  }
  return a;
}

__device__ __forceinline__ v16bf load_b_frag(const bf16_t* base, int ld, int lane) {
  const int n  = lane & 15;
  const int hh = lane >> 4;
  const bf16_t* p = base + (size_t)n * ld + hh * 16;
  v16bf b;
#pragma unroll
  for (int e = 0; e < 16; ++e) b[e] = p[e];
  return b;
}

// ---------------------------------------------------------------------------
// Tensor Data Mover: 2D tile (bf16 elements) global -> LDS.
// D# per CDNA5 ISA. Groups 2/3 zero (2D tensor).
// ---------------------------------------------------------------------------
__device__ __forceinline__ void tdm_load_2d_bf16(unsigned int lds_off, const void* gptr,
                                                 int tensor_d0, int tensor_d1,
                                                 int tile_d0, int tile_d1,
                                                 int stride0) {
  unsigned long long ga = (unsigned long long)(uintptr_t)gptr;
  v4u g0;
  g0[0] = 1u;                                   // count=1, user mode, no gather
  g0[1] = lds_off;                              // lds_addr (bytes)
  g0[2] = (unsigned int)ga;                     // global_addr[31:0]
  g0[3] = (unsigned int)((ga >> 32) & 0x1FFFFFFu) | 0x80000000u; // [56:32] | type=2
  v8i g1;
  g1[0] = 1 << 16;                              // data_size = 1 (2 bytes)
  g1[1] = (tensor_d0 & 0xffff) << 16;           // tensor_dim0[15:0]
  g1[2] = ((tensor_d0 >> 16) & 0xffff) | ((tensor_d1 & 0xffff) << 16);
  g1[3] = ((tensor_d1 >> 16) & 0xffff) | ((tile_d0 & 0xffff) << 16);
  g1[4] = (tile_d1 & 0xffff);                   // tile_dim1 (tile_dim2 = 0)
  g1[5] = stride0;                              // tensor_dim0_stride[31:0]
  g1[6] = 0;
  g1[7] = 0;
  v4i gz = {0, 0, 0, 0};
#if defined(__clang_major__) && __clang_major__ >= 23
  v8i gz8 = {0, 0, 0, 0, 0, 0, 0, 0};
  __builtin_amdgcn_tensor_load_to_lds(g0, g1, gz, gz, gz8, 0);
#else
  __builtin_amdgcn_tensor_load_to_lds(g0, g1, gz, gz, 0);
#endif
}

// ---------------------------------------------------------------------------
// f32 -> bf16 conversion
// ---------------------------------------------------------------------------
__global__ void f32_to_bf16_kernel(const float* __restrict__ src,
                                   bf16_t* __restrict__ dst, int n) {
  int i = blockIdx.x * blockDim.x + threadIdx.x;
  if (i < n) dst[i] = (bf16_t)src[i];
}

// ---------------------------------------------------------------------------
// Generic GEMM: Y = X[M,K] @ W[N,K]^T + bias,  K = N = 1024, M = 4096
// mode 0: bf16 out -> [b, h, s, hd]  (Q, K projections)
// mode 1: bf16 out -> [b, h, hd, s]  (V transposed for PV B-frags)
// mode 2: f32 out [m, n]             (final projection)
// ---------------------------------------------------------------------------
__global__ __launch_bounds__(256)
void gemm_xwt_kernel(const bf16_t* __restrict__ X, const bf16_t* __restrict__ W,
                     const float* __restrict__ bias, void* __restrict__ out,
                     int mode) {
  const int Kd = 1024;
  const int wave = threadIdx.x >> 5;
  const int lane = threadIdx.x & 31;
  const int m0 = blockIdx.x * 128 + (wave & 3) * 32;
  const int n0 = blockIdx.y * 64 + (wave >> 2) * 32;

  v8f c00 = {}, c01 = {}, c10 = {}, c11 = {};

  for (int kt = 0; kt < Kd; kt += 32) {
    v16bf a0 = load_a_frag(X + (size_t)m0 * Kd + kt, Kd, lane);
    v16bf a1 = load_a_frag(X + (size_t)(m0 + 16) * Kd + kt, Kd, lane);
    v16bf b0 = load_b_frag(W + (size_t)n0 * Kd + kt, Kd, lane);
    v16bf b1 = load_b_frag(W + (size_t)(n0 + 16) * Kd + kt, Kd, lane);
    c00 = WMMA_BF16(a0, b0, c00);
    c01 = WMMA_BF16(a0, b1, c01);
    c10 = WMMA_BF16(a1, b0, c10);
    c11 = WMMA_BF16(a1, b1, c11);
  }

  const int hh8 = (lane >> 4) << 3;
  const int nl  = lane & 15;
#pragma unroll
  for (int t = 0; t < 4; ++t) {
    const v8f& c = (t == 0) ? c00 : (t == 1) ? c01 : (t == 2) ? c10 : c11;
    const int mt = m0 + ((t >> 1) << 4);
    const int nt = n0 + ((t & 1) << 4);
    const int n  = nt + nl;
    const float bv = bias[n];
    const int h = n >> 6, hd = n & 63;
#pragma unroll
    for (int v = 0; v < 8; ++v) {
      const int m = mt + v + hh8;
      const float val = c[v] + bv;
      if (mode == 2) {
        ((float*)out)[(size_t)m * Dn + n] = val;
      } else {
        const int b = m >> 11, s = m & (Sn - 1);
        if (mode == 0)
          ((bf16_t*)out)[(((size_t)(b * Hn + h) * Sn + s) << 6) + hd] = (bf16_t)val;
        else
          ((bf16_t*)out)[(((size_t)(b * Hn + h) * HDn + hd) << 11) + s] = (bf16_t)val;
      }
    }
  }
}

// ---------------------------------------------------------------------------
// Flash attention with TDM-staged K/V tiles.
// Per block: 8 waves x 16 queries = 128 queries of one (b,h).
// Key loop in tiles of 32: wave 0 DMAs K tile (32x64) and V tile (64x32)
// into double-buffered LDS with tensor_load_to_lds, overlapped with compute;
// all waves read fragments from LDS. P relayout via per-wave LDS staging.
// ---------------------------------------------------------------------------
struct AttnSmem {
  bf16_t kt[2][32 * 64];   // [key tile buf][32 keys][64 hd]
  bf16_t vt[2][64 * 32];   // [key tile buf][64 hd][32 keys]
  bf16_t p[8][16 * 32];    // per-wave P staging
};

__global__ __launch_bounds__(256)
void attn_kernel(const bf16_t* __restrict__ Q, const bf16_t* __restrict__ Km,
                 const bf16_t* __restrict__ Vt, const int* __restrict__ mask,
                 bf16_t* __restrict__ ctx) {
  __shared__ __align__(16) AttnSmem smem;

  const int wave = threadIdx.x >> 5;
  const int lane = threadIdx.x & 31;
  const int bh = blockIdx.y;            // b*H + h
  const int b = bh >> 4;
  const int h = bh & 15;
  const int q0 = blockIdx.x * 128 + wave * 16;

  const bf16_t* Qbh = Q  + (size_t)bh * Sn * HDn;
  const bf16_t* Kbh = Km + (size_t)bh * Sn * HDn;
  const bf16_t* Vbh = Vt + (size_t)bh * HDn * Sn;
  const int* maskb  = mask + (size_t)b * Sn * Sn;

  // LDS byte offsets for the TDM descriptors (low 32 bits of flat shared
  // address == LDS offset per the aperture mapping).
  const unsigned int ktOff0 = (unsigned int)(uintptr_t)(&smem.kt[0][0]);
  const unsigned int ktOff1 = (unsigned int)(uintptr_t)(&smem.kt[1][0]);
  const unsigned int vtOff0 = (unsigned int)(uintptr_t)(&smem.vt[0][0]);
  const unsigned int vtOff1 = (unsigned int)(uintptr_t)(&smem.vt[1][0]);

  // Q fragments, resident across the key loop (hd 0..31 and 32..63)
  const v16bf aq0 = load_a_frag(Qbh + (size_t)q0 * HDn + 0,  HDn, lane);
  const v16bf aq1 = load_a_frag(Qbh + (size_t)q0 * HDn + 32, HDn, lane);

  v8f o0 = {}, o1 = {}, o2 = {}, o3 = {};
  float mrow[8], lrow[8];
#pragma unroll
  for (int v = 0; v < 8; ++v) { mrow[v] = -1e30f; lrow[v] = 0.f; }

  const float scale = 0.125f;          // 1/sqrt(64)
  const int hh8 = (lane >> 4) << 3;
  const int nl  = lane & 15;
  bf16_t* pl = smem.p[wave];

  // Prologue: DMA tile 0 into buffer 0.
  if (wave == 0) {
    tdm_load_2d_bf16(ktOff0, Kbh, HDn, 32, HDn, 32, HDn);
    tdm_load_2d_bf16(vtOff0, Vbh, 32, HDn, 32, HDn, Sn);
    __builtin_amdgcn_s_wait_tensorcnt(0);
  }
  __syncthreads();

  for (int j = 0; j < Sn / 32; ++j) {
    const int k0 = j * 32;
    const int buf = j & 1;

    // Prefetch next tile into the other buffer (overlaps the wmma work).
    if (wave == 0 && j + 1 < Sn / 32) {
      const int kn = k0 + 32;
      tdm_load_2d_bf16(buf ? ktOff0 : ktOff1, Kbh + (size_t)kn * HDn,
                       HDn, 32, HDn, 32, HDn);
      tdm_load_2d_bf16(buf ? vtOff0 : vtOff1, Vbh + kn,
                       32, HDn, 32, HDn, Sn);
    }

    const bf16_t* kt = smem.kt[buf];
    const bf16_t* vt = smem.vt[buf];

    // ---- scores: 16 q x 32 keys (two C frags) -------------------------
    v8f s0 = {}, s1 = {};
    {
      v16bf kb0 = load_b_frag(kt + 0 * HDn + 0,  HDn, lane);
      v16bf kb1 = load_b_frag(kt + 0 * HDn + 32, HDn, lane);
      s0 = WMMA_BF16(aq0, kb0, s0);
      s0 = WMMA_BF16(aq1, kb1, s0);
      v16bf kb2 = load_b_frag(kt + 16 * HDn + 0,  HDn, lane);
      v16bf kb3 = load_b_frag(kt + 16 * HDn + 32, HDn, lane);
      s1 = WMMA_BF16(aq0, kb2, s1);
      s1 = WMMA_BF16(aq1, kb3, s1);
    }

    // ---- scale + mask -------------------------------------------------
    const int kc0 = k0 + nl;
#pragma unroll
    for (int v = 0; v < 8; ++v) {
      const int q = q0 + v + hh8;
      float v0 = s0[v] * scale;
      float v1 = s1[v] * scale;
      if (maskb[(size_t)q * Sn + kc0] == 0)      v0 = -1e30f;
      if (maskb[(size_t)q * Sn + kc0 + 16] == 0) v1 = -1e30f;
      s0[v] = v0; s1[v] = v1;
    }

    // ---- online softmax (row reductions across 16-lane halves) --------
#pragma unroll
    for (int v = 0; v < 8; ++v) {
      float t = fmaxf(s0[v], s1[v]);
      t = fmaxf(t, __shfl_xor(t, 1, 32));
      t = fmaxf(t, __shfl_xor(t, 2, 32));
      t = fmaxf(t, __shfl_xor(t, 4, 32));
      t = fmaxf(t, __shfl_xor(t, 8, 32));
      const float mn = fmaxf(mrow[v], t);
      const float alpha = __expf(mrow[v] - mn);
      const float p0 = __expf(s0[v] - mn);
      const float p1 = __expf(s1[v] - mn);
      float rs = p0 + p1;
      rs += __shfl_xor(rs, 1, 32);
      rs += __shfl_xor(rs, 2, 32);
      rs += __shfl_xor(rs, 4, 32);
      rs += __shfl_xor(rs, 8, 32);
      lrow[v] = lrow[v] * alpha + rs;
      mrow[v] = mn;
      o0[v] *= alpha; o1[v] *= alpha; o2[v] *= alpha; o3[v] *= alpha;
      s0[v] = p0; s1[v] = p1;
    }

    // ---- P: C-layout -> LDS [q][k] -> A-layout fragment ---------------
#pragma unroll
    for (int v = 0; v < 8; ++v) {
      const int row = v + hh8;
      pl[row * 32 + nl]      = (bf16_t)s0[v];
      pl[row * 32 + 16 + nl] = (bf16_t)s1[v];
    }
    asm volatile("s_wait_dscnt 0" ::: "memory");
    v16bf ap = load_a_frag(pl, 32, lane);

    // ---- O += P(16x32 keys) x V(32 keys x 64 hd) ----------------------
    v16bf vb0 = load_b_frag(vt + 0  * 32, 32, lane);
    v16bf vb1 = load_b_frag(vt + 16 * 32, 32, lane);
    v16bf vb2 = load_b_frag(vt + 32 * 32, 32, lane);
    v16bf vb3 = load_b_frag(vt + 48 * 32, 32, lane);
    o0 = WMMA_BF16(ap, vb0, o0);
    o1 = WMMA_BF16(ap, vb1, o1);
    o2 = WMMA_BF16(ap, vb2, o2);
    o3 = WMMA_BF16(ap, vb3, o3);

    // Publish the prefetched buffer for the next iteration.
    if (wave == 0 && j + 1 < Sn / 32) __builtin_amdgcn_s_wait_tensorcnt(0);
    __syncthreads();
  }

  // ---- finalize: O /= l, store ctx[b][q][h*64+hd] bf16 -----------------
#pragma unroll
  for (int v = 0; v < 8; ++v) {
    const float inv = 1.f / lrow[v];
    const int q = q0 + v + hh8;
    const size_t base = ((size_t)b * Sn + q) * Dn + h * HDn;
    ctx[base + 0  + nl] = (bf16_t)(o0[v] * inv);
    ctx[base + 16 + nl] = (bf16_t)(o1[v] * inv);
    ctx[base + 32 + nl] = (bf16_t)(o2[v] * inv);
    ctx[base + 48 + nl] = (bf16_t)(o3[v] * inv);
  }
}

// ---------------------------------------------------------------------------
// Launch
// ---------------------------------------------------------------------------
extern "C" void kernel_launch(void* const* d_in, const int* in_sizes, int n_in,
                              void* d_out, int out_size, void* d_ws, size_t ws_size,
                              hipStream_t stream) {
  const float* x    = (const float*)d_in[0];
  const int*   mask = (const int*)d_in[1];
  const float* Wq   = (const float*)d_in[2];
  const float* bq   = (const float*)d_in[3];
  const float* Wk   = (const float*)d_in[4];
  const float* bk   = (const float*)d_in[5];
  const float* Wv   = (const float*)d_in[6];
  const float* bv   = (const float*)d_in[7];
  const float* Wo   = (const float*)d_in[8];
  const float* bo   = (const float*)d_in[9];
  float* out = (float*)d_out;

  char* ws = (char*)d_ws;
  size_t off = 0;
  auto salloc = [&](size_t bytes) -> void* {
    void* p = ws + off;
    off += (bytes + 255) & ~(size_t)255;
    return p;
  };
  bf16_t* xb   = (bf16_t*)salloc((size_t)Mn * Dn * 2);
  bf16_t* wqb  = (bf16_t*)salloc((size_t)Dn * Dn * 2);
  bf16_t* wkb  = (bf16_t*)salloc((size_t)Dn * Dn * 2);
  bf16_t* wvb  = (bf16_t*)salloc((size_t)Dn * Dn * 2);
  bf16_t* wob  = (bf16_t*)salloc((size_t)Dn * Dn * 2);
  bf16_t* Qb   = (bf16_t*)salloc((size_t)Mn * Dn * 2);   // [b,h,s,hd]
  bf16_t* Kb   = (bf16_t*)salloc((size_t)Mn * Dn * 2);   // [b,h,s,hd]
  bf16_t* Vtb  = (bf16_t*)salloc((size_t)Mn * Dn * 2);   // [b,h,hd,s]
  bf16_t* ctxb = (bf16_t*)salloc((size_t)Mn * Dn * 2);   // [b,s,d]

  const int nX = Mn * Dn;
  const int nW = Dn * Dn;
  f32_to_bf16_kernel<<<nX / 256, 256, 0, stream>>>(x,  xb,  nX);
  f32_to_bf16_kernel<<<nW / 256, 256, 0, stream>>>(Wq, wqb, nW);
  f32_to_bf16_kernel<<<nW / 256, 256, 0, stream>>>(Wk, wkb, nW);
  f32_to_bf16_kernel<<<nW / 256, 256, 0, stream>>>(Wv, wvb, nW);
  f32_to_bf16_kernel<<<nW / 256, 256, 0, stream>>>(Wo, wob, nW);

  dim3 gg(Mn / 128, Dn / 64);
  gemm_xwt_kernel<<<gg, 256, 0, stream>>>(xb, wqb, bq, Qb,  0);
  gemm_xwt_kernel<<<gg, 256, 0, stream>>>(xb, wkb, bk, Kb,  0);
  gemm_xwt_kernel<<<gg, 256, 0, stream>>>(xb, wvb, bv, Vtb, 1);

  attn_kernel<<<dim3(Sn / 128, Bn * Hn), 256, 0, stream>>>(Qb, Kb, Vtb, mask, ctxb);

  gemm_xwt_kernel<<<gg, 256, 0, stream>>>(ctxb, wob, bo, out, 2);
}
